// GCNEncoder_25486335934641
// MI455X (gfx1250) — compile-verified, matrix operand order checked by reading, not run on hardware
//
#include <hip/hip_runtime.h>
#include <hip/hip_bf16.h>

typedef __attribute__((ext_vector_type(16))) __bf16 v16bf;
typedef __attribute__((ext_vector_type(8)))  float  v8f;

union FragU {
    v16bf v;
    unsigned short u16[16];
    uint4 u4[2];
};

__device__ __forceinline__ unsigned short f2bf(float f) {
    unsigned int u = __float_as_uint(f);
    unsigned int r = u + 0x7FFFu + ((u >> 16) & 1u);   // round-to-nearest-even
    return (unsigned short)(r >> 16);
}

// ---------------------------------------------------------------- degrees ---
__global__ void k_deg_init(float* deg, int n) {
    int i = blockIdx.x * blockDim.x + threadIdx.x;
    if (i < n) deg[i] = 1.0f;                          // self-loop
}

__global__ void k_deg_accum(const int* dst, float* deg, int e) {
    int i = blockIdx.x * blockDim.x + threadIdx.x;
    if (i < e) unsafeAtomicAdd(&deg[dst[i]], 1.0f);
}

__global__ void k_deg_finish(float* deg, int n) {
    int i = blockIdx.x * blockDim.x + threadIdx.x;
    if (i < n) deg[i] = rsqrtf(deg[i]);                // deg >= 1 always
}

// ------------------------------------------------------------- conversions --
__global__ void k_cvt_bf16(const float* in, unsigned short* out, int n) {
    int i = blockIdx.x * blockDim.x + threadIdx.x;
    if (i < n) out[i] = f2bf(in[i]);
}

// Pack W [K x N] row-major fp32 into per-lane WMMA B fragments (bf16).
// Tile (kt, nt): lane n<16 -> col nt*16+n, K = kt*32 + {0..7, 16..23};
// lanes 16..31  -> col nt*16+n-16, K = kt*32 + {8..15, 24..31}.
// One wave (32 threads) per tile; 16 ushorts per lane, contiguous.
__global__ void k_pack_w(const float* W, unsigned short* out, int K, int N) {
    int tile = blockIdx.x;
    int lane = threadIdx.x;                            // 0..31
    int ntiles = N >> 4;
    int kt = tile / ntiles, nt = tile - kt * ntiles;
    int col = nt * 16 + (lane & 15);
    int kb  = kt * 32 + ((lane < 16) ? 0 : 8);
    unsigned short* o = out + (size_t)tile * 512 + lane * 16;
#pragma unroll
    for (int e = 0; e < 8; ++e) o[e]     = f2bf(W[(kb + e)      * N + col]);
#pragma unroll
    for (int e = 0; e < 8; ++e) o[8 + e] = f2bf(W[(kb + 16 + e) * N + col]);
}

// ------------------------------------------------------------------- GEMM ---
// C[M x 128] = A[M x K](bf16, row-major) * Bpacked[K/32 x 8 tiles].
// B fragments are staged once per block into LDS (<= 64 KB, dynamic),
// shared by all 8 waves; WMMAs read B via ds_load_b128.
// One wave per 16-row strip; 8 N-tiles accumulated in 8 v8f frags.
__global__ void k_gemm_bf16(const unsigned short* __restrict__ A,
                            const unsigned short* __restrict__ Bp,
                            float* __restrict__ C, int M, int K) {
    extern __shared__ unsigned short sB[];             // (K/32)*8*512 ushorts

    // cooperative global -> LDS stage of all B fragments (uint4 granularity)
    {
        const uint4* gB  = (const uint4*)Bp;
        uint4*       sB4 = (uint4*)sB;
        int n4 = (K >> 5) * 512;                       // (K/32)*8*512/8
        for (int i = threadIdx.x; i < n4; i += blockDim.x) sB4[i] = gB[i];
    }
    __syncthreads();

    int wave   = (int)((blockIdx.x * blockDim.x + threadIdx.x) >> 5);
    int lane   = threadIdx.x & 31;
    int mtiles = (M + 15) >> 4;
    if (wave >= mtiles) return;

    int row = wave * 16 + (lane & 15);
    if (row >= M) row = M - 1;                         // safe duplicate load
    int kb0 = (lane < 16) ? 0 : 8;
    const unsigned short* arow = A + (size_t)row * K;

    v8f zero = {};
    v8f acc[8];
#pragma unroll
    for (int nt = 0; nt < 8; ++nt) acc[nt] = zero;

    for (int kt = 0; kt < K; kt += 32) {
        FragU a;
        a.u4[0] = *(const uint4*)(arow + kt + kb0);
        a.u4[1] = *(const uint4*)(arow + kt + kb0 + 16);
        const unsigned short* bp = sB + (size_t)(kt >> 5) * 8 * 512 + lane * 16;
#pragma unroll
        for (int nt = 0; nt < 8; ++nt) {
            FragU b;
            const uint4* bl = (const uint4*)(bp + nt * 512);
            b.u4[0] = bl[0];
            b.u4[1] = bl[1];
            acc[nt] = __builtin_amdgcn_wmma_f32_16x16x32_bf16(
                false, a.v, false, b.v, (short)0, acc[nt], false, false);
        }
    }

    int srow0 = wave * 16 + ((lane < 16) ? 0 : 8);
    int scol  = lane & 15;
#pragma unroll
    for (int nt = 0; nt < 8; ++nt) {
#pragma unroll
        for (int j = 0; j < 8; ++j) {
            int r = srow0 + j;
            if (r < M) C[(size_t)r * 128 + nt * 16 + scol] = acc[nt][j];
        }
    }
}

// ------------------------------------------------------------ aggregation ---
__global__ void k_agg_init(const float* h, const float* dinv, float* agg, int total) {
    int i = blockIdx.x * blockDim.x + threadIdx.x;
    if (i < total) {
        float d = dinv[i >> 7];
        agg[i] = d * d * h[i];                         // self-loop message
    }
}

// One wave per edge: gather h[src] (float4/lane), atomic scatter into agg[dst].
__global__ void k_agg_edges(const int* __restrict__ src, const int* __restrict__ dst,
                            const float* __restrict__ dinv,
                            const float* __restrict__ h,
                            float* __restrict__ agg, int e) {
    int wid  = (int)((blockIdx.x * blockDim.x + threadIdx.x) >> 5);
    if (wid >= e) return;
    int lane = threadIdx.x & 31;
    int s = src[wid], d = dst[wid];
    float nrm = dinv[s] * dinv[d];
    const float4 hv = *(const float4*)(h + (size_t)s * 128 + lane * 4);
    float* ap = agg + (size_t)d * 128 + lane * 4;
    unsafeAtomicAdd(ap + 0, nrm * hv.x);
    unsafeAtomicAdd(ap + 1, nrm * hv.y);
    unsafeAtomicAdd(ap + 2, nrm * hv.z);
    unsafeAtomicAdd(ap + 3, nrm * hv.w);
}

// ------------------------------------------- bias + LayerNorm + ELU (+res) --
// One wave per node row of 128. mode 0: write f32 + bf16 copy (layer 1).
// mode 1: add residual, write final f32 output (layer 2).
__global__ void k_ln_elu(const float* __restrict__ agg,
                         const float* __restrict__ bias,
                         const float* __restrict__ gamma,
                         const float* __restrict__ beta,
                         const float* __restrict__ res_in,
                         float* __restrict__ out_f32,
                         unsigned short* __restrict__ out_bf16,
                         int n, int mode) {
    int wave = (int)((blockIdx.x * blockDim.x + threadIdx.x) >> 5);
    if (wave >= n) return;
    int lane = threadIdx.x & 31;
    const float* a = agg + (size_t)wave * 128;

    float v[4];
    float s = 0.f;
#pragma unroll
    for (int j = 0; j < 4; ++j) {
        int c = lane * 4 + j;
        v[j] = a[c] + bias[c];
        s += v[j];
    }
#pragma unroll
    for (int off = 16; off >= 1; off >>= 1) s += __shfl_xor(s, off, 32);
    float mu = s * (1.0f / 128.0f);

    float q = 0.f;
#pragma unroll
    for (int j = 0; j < 4; ++j) {
        float dv = v[j] - mu;
        q += dv * dv;
    }
#pragma unroll
    for (int off = 16; off >= 1; off >>= 1) q += __shfl_xor(q, off, 32);
    float inv = rsqrtf(q * (1.0f / 128.0f) + 1e-5f);

#pragma unroll
    for (int j = 0; j < 4; ++j) {
        int c = lane * 4 + j;
        float y = (v[j] - mu) * inv * gamma[c] + beta[c];
        y = (y > 0.f) ? y : expm1f(y);                 // ELU, alpha=1
        size_t idx = (size_t)wave * 128 + c;
        if (mode == 0) {
            out_f32[idx]  = y;                         // residual source
            out_bf16[idx] = f2bf(y);                   // GEMM-2 input
        } else {
            out_f32[idx] = y + res_in[idx];            // final output
        }
    }
}

// ------------------------------------------------------------------ driver --
extern "C" void kernel_launch(void* const* d_in, const int* in_sizes, int n_in,
                              void* d_out, int out_size, void* d_ws, size_t ws_size,
                              hipStream_t stream) {
    const float* x   = (const float*)d_in[0];
    const float* W1  = (const float*)d_in[1];
    const float* b1  = (const float*)d_in[2];
    const float* g1  = (const float*)d_in[3];
    const float* be1 = (const float*)d_in[4];
    const float* W2  = (const float*)d_in[5];
    const float* b2  = (const float*)d_in[6];
    const float* g2  = (const float*)d_in[7];
    const float* be2 = (const float*)d_in[8];
    const int*   ei  = (const int*)d_in[9];

    const int EMB = in_sizes[2];                       // 128
    const int IN  = in_sizes[1] / EMB;                 // 256
    const int N   = in_sizes[0] / IN;                  // 50000
    const int E   = in_sizes[9] / 2;                   // 800000
    const int* src = ei;
    const int* dst = ei + E;

    // workspace layout (256B aligned)
    char* base = (char*)d_ws;
    size_t off = 0;
    auto alloc = [&](size_t bytes) {
        char* p = base + off;
        off += (bytes + 255) & ~(size_t)255;
        return p;
    };
    float*          dinv  = (float*)alloc((size_t)N * 4);
    unsigned short* xb    = (unsigned short*)alloc((size_t)N * IN * 2);
    unsigned short* W1p   = (unsigned short*)alloc((size_t)IN * EMB * 2);
    unsigned short* W2p   = (unsigned short*)alloc((size_t)EMB * EMB * 2);
    float*          h     = (float*)alloc((size_t)N * EMB * 4);
    float*          agg   = (float*)alloc((size_t)N * EMB * 4);
    float*          res   = (float*)alloc((size_t)N * EMB * 4);
    unsigned short* h1b   = (unsigned short*)alloc((size_t)N * EMB * 2);
    (void)ws_size;

    const int B = 256;
    int total = N * EMB;

    // degrees -> dinv
    k_deg_init  <<<(N + B - 1) / B, B, 0, stream>>>(dinv, N);
    k_deg_accum <<<(E + B - 1) / B, B, 0, stream>>>(dst, dinv, E);
    k_deg_finish<<<(N + B - 1) / B, B, 0, stream>>>(dinv, N);

    // conversions / packing
    k_cvt_bf16<<<((size_t)N * IN + B - 1) / B, B, 0, stream>>>(x, xb, N * IN);
    k_pack_w  <<<(IN  / 32) * (EMB / 16), 32, 0, stream>>>(W1, W1p, IN,  EMB);
    k_pack_w  <<<(EMB / 32) * (EMB / 16), 32, 0, stream>>>(W2, W2p, EMB, EMB);

    int mtiles = (N + 15) / 16;
    int gemm_blocks = (mtiles + 7) / 8;                // 8 waves/block
    size_t lds1 = (size_t)(IN  / 32) * 8 * 1024;       // bytes of packed B (64 KB)
    size_t lds2 = (size_t)(EMB / 32) * 8 * 1024;       // 32 KB

    // ---- layer 1 ----
    k_gemm_bf16<<<gemm_blocks, B, lds1, stream>>>(xb, W1p, h, N, IN);
    k_agg_init <<<(total + B - 1) / B, B, 0, stream>>>(h, dinv, agg, total);
    k_agg_edges<<<(E * 32 + B - 1) / B, B, 0, stream>>>(src, dst, dinv, h, agg, E);
    k_ln_elu   <<<(N + 7) / 8, B, 0, stream>>>(agg, b1, g1, be1, nullptr, res, h1b, N, 0);

    // ---- layer 2 ----
    k_gemm_bf16<<<gemm_blocks, B, lds2, stream>>>(h1b, W2p, h, N, EMB);
    k_agg_init <<<(total + B - 1) / B, B, 0, stream>>>(h, dinv, agg, total);
    k_agg_edges<<<(E * 32 + B - 1) / B, B, 0, stream>>>(src, dst, dinv, h, agg, E);
    k_ln_elu   <<<(N + 7) / 8, B, 0, stream>>>(agg, b2, g2, be2, res, (float*)d_out, nullptr, N, 1);
}
